// MPNEncoder_33517924778171
// MI455X (gfx1250) — matrix operand order repaired and checked by reading, not compiled, not run on hardware
//
#include <hip/hip_runtime.h>

typedef __attribute__((ext_vector_type(16))) _Float16 v16h;
typedef __attribute__((ext_vector_type(8)))  _Float16 v8h;
typedef __attribute__((ext_vector_type(8)))  float    v8f;

static constexpr int HID   = 256;   // hidden width
static constexpr int DEG   = 6;     // max degree
static constexpr int NMOLS = 5000;

// ---------------- conversion kernels ----------------

// src [rows][cols] f32  ->  dst [rows][colsPad] f16 (zero-padded K)
__global__ void k_cvt_pad(const float* __restrict__ src, _Float16* __restrict__ dst,
                          int rows, int cols, int colsPad) {
  int i = blockIdx.x * blockDim.x + threadIdx.x;
  int total = rows * colsPad;
  if (i >= total) return;
  int r = i / colsPad, c = i - r * colsPad;
  dst[i] = (c < cols) ? (_Float16)src[(size_t)r * cols + c] : (_Float16)0.0f;
}

// W [K][256] f32  ->  Wt [256][Kpad] f16 (transposed, zero-padded K)
__global__ void k_cvt_wt(const float* __restrict__ W, _Float16* __restrict__ Wt,
                         int K, int Kpad) {
  int i = blockIdx.x * blockDim.x + threadIdx.x;
  int total = HID * Kpad;
  if (i >= total) return;
  int n = i / Kpad, k = i - n * Kpad;
  Wt[i] = (k < K) ? (_Float16)W[(size_t)k * HID + n] : (_Float16)0.0f;
}

// ---------------- WMMA GEMM:  C[M][256] = A[M][Kpad] * Wt^T + bias ----------------
// Block = 128 threads = 4 waves; block handles 16 rows x 256 cols.
// Wave w handles columns [64w, 64w+64): 4 WMMA 16x16 tiles per K-step, A reused.
__global__ __launch_bounds__(128) void k_gemm(const _Float16* __restrict__ A,
    const _Float16* __restrict__ Bt,   // [256][Kpad] (column-major weights)
    const float* __restrict__ bias,    // [256]
    float* __restrict__ C32,           // optional f32 output (nullable)
    _Float16* __restrict__ C16,        // optional f16 output (nullable)
    int Kpad, int relu) {
  const int lane = threadIdx.x & 31;
  const int wave = threadIdx.x >> 5;
  const int m0 = blockIdx.x << 4;
  const int n0 = wave << 6;
  const int hl = lane & 15;
  const bool hi = lane >= 16;

  // A fragment: lane<16 -> row m0+lane, K {0..15};
  //             lane>=16 -> row m0+lane-16, K {16..31}
  const _Float16* Arow = A + (size_t)(m0 + hl) * Kpad + (hi ? 16 : 0);
  // B fragment: lane<16 -> col n0+lane, K k0..k0+15; lane>=16 -> K k0+16..k0+31
  const int boff = hi ? 16 : 0;

  v8f acc[4] = {};
  for (int k0 = 0; k0 < Kpad; k0 += 32) {
    v8h alo = *(const v8h*)(Arow + k0);
    v8h ahi = *(const v8h*)(Arow + k0 + 8);
    v16h a;
#pragma unroll
    for (int i = 0; i < 8; ++i) { a[i] = alo[i]; a[i + 8] = ahi[i]; }
#pragma unroll
    for (int t = 0; t < 4; ++t) {
      const v16h b = *(const v16h*)(Bt + (size_t)(n0 + t * 16 + hl) * Kpad + k0 + boff);
      acc[t] = __builtin_amdgcn_wmma_f32_16x16x32_f16(
          false, a, false, b, (short)0, acc[t], false, false);
    }
  }

  // C/D layout: VGPR r -> (row m0+r, lanes 0..15) / (row m0+8+r, lanes 16..31)
  const int rbase = hi ? 8 : 0;
#pragma unroll
  for (int t = 0; t < 4; ++t) {
    const int ncol = n0 + t * 16 + hl;
    const float bv = bias[ncol];
#pragma unroll
    for (int r = 0; r < 8; ++r) {
      float v = acc[t][r] + bv;
      if (relu) v = fmaxf(v, 0.0f);
      size_t off = (size_t)(m0 + rbase + r) * HID + ncol;
      if (C32) C32[off] = v;
      if (C16) C16[off] = (_Float16)v;
    }
  }
}

// ---------------- GAT attention + residual + relu (wave per atom) ----------------
// h_atoms / h_bonds are f16 (halves gather traffic); math in fp32.
__device__ __forceinline__ float reduce8(float v) {
  v += __shfl_xor(v, 1, 32);
  v += __shfl_xor(v, 2, 32);
  v += __shfl_xor(v, 4, 32);
  return v;
}

__global__ __launch_bounds__(256) void k_attn(const _Float16* __restrict__ hA,
    const _Float16* __restrict__ hB, const int* __restrict__ a2a,
    const int* __restrict__ a2b, const float* __restrict__ asrc,
    const float* __restrict__ adst, const float* __restrict__ aedge,
    float* __restrict__ x32, _Float16* __restrict__ x16, int N) {
  const int lane = threadIdx.x & 31;
  const int n = blockIdx.x * 8 + (threadIdx.x >> 5);
  if (n >= N) return;
  const int d0 = lane * 8;   // lane owns 8 dims; head = lane/8 (hd = 64)

  // fetch all indices, then fire all gathers back-to-back (deep MLP)
  int ia[DEG], ib[DEG];
#pragma unroll
  for (int k = 0; k < DEG; ++k) { ia[k] = a2a[n * DEG + k]; ib[k] = a2b[n * DEG + k]; }

  const v8h selfh = *(const v8h*)(hA + (size_t)n * HID + d0);
  v8h nav[DEG], nbv[DEG];
#pragma unroll
  for (int k = 0; k < DEG; ++k) nav[k] = *(const v8h*)(hA + (size_t)ia[k] * HID + d0);
#pragma unroll
  for (int k = 0; k < DEG; ++k) nbv[k] = *(const v8h*)(hB + (size_t)ib[k] * HID + d0);

  float asv[8], adv[8], aev[8];
#pragma unroll
  for (int j = 0; j < 8; ++j) {
    asv[j] = asrc[d0 + j];
    adv[j] = adst[d0 + j];
    aev[j] = aedge[d0 + j];
  }
  float p = 0.f;
#pragma unroll
  for (int j = 0; j < 8; ++j) p += (float)selfh[j] * asv[j];
  const float ssrc = reduce8(p);

  float val[DEG][8];
  float sc[DEG];
#pragma unroll
  for (int k = 0; k < DEG; ++k) {
    float pd = 0.f, pe = 0.f;
#pragma unroll
    for (int j = 0; j < 8; ++j) {
      const float na = (float)nav[k][j], nb = (float)nbv[k][j];
      val[k][j] = na + nb;
      pd += na * adv[j];
      pe += nb * aev[j];
    }
    float s = ssrc + reduce8(pd) + reduce8(pe);
    s = (s > 0.f) ? s : 0.2f * s;            // leaky relu
    sc[k] = (ia[k] != 0) ? s : -1e9f;        // index 0 == padding atom
  }
  float m = sc[0];
#pragma unroll
  for (int k = 1; k < DEG; ++k) m = fmaxf(m, sc[k]);
  float den = 0.f, al[DEG];
#pragma unroll
  for (int k = 0; k < DEG; ++k) { al[k] = __expf(sc[k] - m); den += al[k]; }
  const float inv = 1.0f / den;

  float out[8] = {};
#pragma unroll
  for (int k = 0; k < DEG; ++k)
#pragma unroll
    for (int j = 0; j < 8; ++j) out[j] += al[k] * val[k][j];

#pragma unroll
  for (int j = 0; j < 8; ++j) {
    const size_t off = (size_t)n * HID + d0 + j;
    const float v = fmaxf(out[j] * inv + x32[off], 0.0f);   // residual + relu
    x32[off] = v;
    x16[off] = (_Float16)v;
  }
}

// ---------------- deterministic segment mean (mol_ids sorted) ----------------
__global__ void k_pool_mean(const float* __restrict__ ah, const int* __restrict__ mol,
                            float* __restrict__ out, int N) {
  int i = blockIdx.x * blockDim.x + threadIdx.x;
  if (i >= NMOLS * HID) return;
  const int m = i >> 8;
  const int c = i & 255;
  int lo = 0, hi = N;
  while (lo < hi) { int mid = (lo + hi) >> 1; if (mol[mid] < m) lo = mid + 1; else hi = mid; }
  const int start = lo;
  int lo2 = start, hi2 = N;
  while (lo2 < hi2) { int mid = (lo2 + hi2) >> 1; if (mol[mid] < m + 1) lo2 = mid + 1; else hi2 = mid; }
  const int end = lo2;
  float s = 0.f;
  for (int a = start; a < end; ++a) s += ah[(size_t)a * HID + c];
  out[i] = s / fmaxf((float)(end - start), 1.0f);
}

// ---------------- host orchestration ----------------
extern "C" void kernel_launch(void* const* d_in, const int* in_sizes, int n_in,
                              void* d_out, int out_size, void* d_ws, size_t ws_size,
                              hipStream_t stream) {
  (void)n_in; (void)out_size; (void)ws_size;
  const float* f_atoms = (const float*)d_in[0];
  const float* f_bonds = (const float*)d_in[1];
  const int*   a2a     = (const int*)d_in[2];
  const int*   a2b     = (const int*)d_in[3];
  const int*   mol_ids = (const int*)d_in[4];
  const float* W_i     = (const float*)d_in[5];
  const float* b_i     = (const float*)d_in[6];
  const float* W_atom  = (const float*)d_in[7];
  const float* b_atom  = (const float*)d_in[8];
  const float* W_bond  = (const float*)d_in[9];
  const float* b_bond  = (const float*)d_in[10];
  const float* att_src = (const float*)d_in[11];
  const float* att_dst = (const float*)d_in[12];
  const float* att_edge= (const float*)d_in[13];
  const float* W_o     = (const float*)d_in[14];
  const float* b_o     = (const float*)d_in[15];

  const int AF = 133, BF = 147;
  const int AFp = 160, BFp = 160;       // K padded to multiples of 32
  const int N = in_sizes[0] / AF;       // 60000 (divisible by 16)
  const int M = in_sizes[1] / BF;       // 120000 (divisible by 16)

  char* ws = (char*)d_ws;
  size_t off = 0;
  auto take = [&](size_t bytes) -> void* {
    size_t o = (off + 255) & ~(size_t)255;
    off = o + bytes;
    return (void*)(ws + o);
  };
  _Float16* fa16 = (_Float16*)take((size_t)N * AFp * 2);
  _Float16* fb16 = (_Float16*)take((size_t)M * BFp * 2);
  _Float16* Wi_t = (_Float16*)take((size_t)HID * AFp * 2);
  _Float16* Wa_t = (_Float16*)take((size_t)3 * HID * HID * 2);
  _Float16* Wb_t = (_Float16*)take((size_t)3 * HID * BFp * 2);
  _Float16* Wo_t = (_Float16*)take((size_t)HID * HID * 2);
  float*    x32  = (float*)   take((size_t)N * HID * 4);
  _Float16* x16  = (_Float16*)take((size_t)N * HID * 2);
  _Float16* hA16 = (_Float16*)take((size_t)N * HID * 2);   // h_atoms (f16 gathers)
  _Float16* hB16 = (_Float16*)take((size_t)M * HID * 2);   // h_bonds (f16 gathers)
  float*    ah32 = (float*)   take((size_t)N * HID * 4);   // final atom_hiddens

  const int TPB = 256;
  // f32 -> padded f16 inputs, transposed f16 weights
  k_cvt_pad<<<(N * AFp + TPB - 1) / TPB, TPB, 0, stream>>>(f_atoms, fa16, N, AF, AFp);
  k_cvt_pad<<<(M * BFp + TPB - 1) / TPB, TPB, 0, stream>>>(f_bonds, fb16, M, BF, BFp);
  k_cvt_wt<<<(HID * AFp + TPB - 1) / TPB, TPB, 0, stream>>>(W_i, Wi_t, AF, AFp);
  for (int d = 0; d < 3; ++d) {
    k_cvt_wt<<<(HID * HID + TPB - 1) / TPB, TPB, 0, stream>>>(
        W_atom + (size_t)d * HID * HID, Wa_t + (size_t)d * HID * HID, HID, HID);
    k_cvt_wt<<<(HID * BFp + TPB - 1) / TPB, TPB, 0, stream>>>(
        W_bond + (size_t)d * BF * HID, Wb_t + (size_t)d * HID * BFp, BF, BFp);
  }
  k_cvt_wt<<<(HID * HID + TPB - 1) / TPB, TPB, 0, stream>>>(W_o, Wo_t, HID, HID);

  // x = relu(f_atoms @ W_i + b_i)   (need f32 for residual + f16 for next GEMM)
  k_gemm<<<N / 16, 128, 0, stream>>>(fa16, Wi_t, b_i, x32, x16, AFp, 1);

  for (int d = 0; d < 3; ++d) {
    // h_atoms / h_bonds: f16 only (gather traffic halved)
    k_gemm<<<N / 16, 128, 0, stream>>>(x16, Wa_t + (size_t)d * HID * HID,
                                       b_atom + d * HID, (float*)nullptr, hA16, HID, 0);
    k_gemm<<<M / 16, 128, 0, stream>>>(fb16, Wb_t + (size_t)d * HID * BFp,
                                       b_bond + d * HID, (float*)nullptr, hB16, BFp, 0);
    k_attn<<<(N + 7) / 8, 256, 0, stream>>>(hA16, hB16, a2a, a2b,
                                            att_src + d * HID, att_dst + d * HID,
                                            att_edge + d * HID, x32, x16, N);
  }

  // atom_hiddens = x @ W_o + b_o   (f32 for pooling)
  k_gemm<<<N / 16, 128, 0, stream>>>(x16, Wo_t, b_o, ah32, (_Float16*)nullptr, HID, 0);

  // deterministic per-molecule mean pool
  k_pool_mean<<<(NMOLS * HID + TPB - 1) / TPB, TPB, 0, stream>>>(ah32, mol_ids, (float*)d_out, N);
}
